// CritiGraph_53420803227703
// MI455X (gfx1250) — compile-verified, matrix operand order checked by reading, not run on hardware
//
#include <hip/hip_runtime.h>
#include <hip/hip_bf16.h>

#define T_CNT   512
#define NC      65
#define TPX     8
#define N_TT    64
#define K_VOC   128
#define S_TOT   192
#define D_DIM   256
#define NCOL    (NC * TPX)   // 520

typedef __attribute__((ext_vector_type(16))) _Float16     v16h;
typedef __attribute__((ext_vector_type(4)))  _Float16     v4h;
typedef __attribute__((ext_vector_type(8)))  float        v8f;
typedef __attribute__((ext_vector_type(4)))  float        v4f;
typedef __attribute__((ext_vector_type(4)))  unsigned int v4u;

union V16H { v16h v; v4u q[2]; };

// CritiGraph frexp-cosine: sgn(a)*sgn(b) * (1 - bitlen(|a|^|b| + 1)/16)
__device__ __forceinline__ float cg_cos(int c1, int c2) {
    float sg = ((c1 ^ c2) >= 0) ? 1.0f : -1.0f;   // sign bits equal <=> product +1
    int a1 = c1 < 0 ? -c1 : c1;
    int a2 = c2 < 0 ? -c2 : c2;
    int e  = 32 - __clz((a1 ^ a2) + 1);           // == frexp exponent (exact, v <= 2^16)
    return sg * (1.0f - (float)e * 0.0625f);
}

__global__ __launch_bounds__(256, 2)
void critigraph_kernel(const int* __restrict__ cur_tar,
                       const int* __restrict__ cnc_loc,
                       const int* __restrict__ sta_loc,
                       const int* __restrict__ ttn_loc,
                       const int* __restrict__ voc_loc,
                       const float* __restrict__ sta_emb,
                       const float* __restrict__ ttn_emb,
                       float* __restrict__ out_dd,
                       float* __restrict__ out_ds) {
    const int t    = blockIdx.x;
    const int tid  = threadIdx.x;
    const int lane = tid & 31;                    // wave32
    const int wave = tid >> 5;

    __shared__ int   sPos[S_TOT * TPX];           // pos_loc = [ttn_loc ; voc_loc]
    __shared__ int   sCnc[NCOL];
    __shared__ int   sSta[TPX];
    __shared__ float sCosSp[S_TOT * TPX];
    __shared__ float sCosSum[S_TOT];
    __shared__ float sEu[N_TT];
    __shared__ __align__(16) _Float16 sStaH[D_DIM];
    __shared__ __align__(16) _Float16 sStage[4][16 * 32];  // per-wave f16 B tile
    __shared__ float sPeu[N_TT];
    __shared__ float sK0;
    __shared__ int   sTar;

    // ---- phase 0: stage loc tables + sta_emb (as f16) into LDS ----
    for (int i = tid; i < N_TT * TPX;  i += 256) sPos[i]              = ttn_loc[t * N_TT * TPX + i];
    for (int i = tid; i < K_VOC * TPX; i += 256) sPos[N_TT * TPX + i] = voc_loc[t * K_VOC * TPX + i];
    for (int i = tid; i < NCOL;        i += 256) sCnc[i]              = cnc_loc[t * NCOL + i];
    if (tid < TPX) sSta[tid] = sta_loc[t * TPX + tid];
    sStaH[tid] = (_Float16)sta_emb[t * D_DIM + tid];      // 256 threads == 256 elems
    if (tid == 0) sTar = cur_tar[t];
    __syncthreads();

    // ---- phase 1: cos_sp[s][p], cos_sum[s] (threads 0..191, whole waves) ----
    if (tid < S_TOT) {
        float sum = 0.f;
        #pragma unroll
        for (int p = 0; p < TPX; ++p) {
            float v = cg_cos(sSta[p], sPos[tid * TPX + p]);
            sCosSp[tid * TPX + p] = v;
            sum += v;
        }
        sCosSum[tid] = sum;
    }

    // ---- phase 1b: eu[0:64] = sta_emb . ttn_emb[s]  via v_wmma_f32_16x16x32_f16 ----
    // (eu_val[:, N_TTNBR:] is never used by the reference -> voc_emb never touched)
    for (int tile = wave; tile < 4; tile += 8) {          // waves 0..3, uniform per wave
        const int sBase = tile * 16;
        const float* src = ttn_emb + ((size_t)t * N_TT + sBase) * D_DIM;
        _Float16* stg = &sStage[wave][0];
        const int base0 = (lane & 16) ? 8 : 0;            // K-half select per ISA A/B layout
        const int sl = lane & 15;
        v8f acc = {};
        for (int ch = 0; ch < 8; ++ch) {                  // K chunks of 32 along D
            const int dBase = ch * 32;
            #pragma unroll
            for (int it = 0; it < 4; ++it) {              // 16 rows x 32 f32 -> f16 tile
                int f   = it * 32 + lane;
                int row = f >> 3;
                int col = (f & 7) << 2;
                v4f x = *(const v4f*)(src + row * D_DIM + dBase + col);
                v4h h = { (_Float16)x.x, (_Float16)x.y, (_Float16)x.z, (_Float16)x.w };
                *(v4h*)(stg + row * 32 + col) = h;
            }
            if (ch < 7)
                __builtin_prefetch(src + (lane >> 3) * D_DIM + dBase + 32, 0, 1);
            __builtin_amdgcn_wave_barrier();
            asm volatile("s_wait_dscnt 0" ::: "memory");  // cross-lane RAW through LDS
            V16H a, b;
            // B[k, n=sl]: halves j<8 -> K=base0+j ; j>=8 -> K=16+base0+(j-8)
            b.q[0] = *(const v4u*)(stg + sl * 32 + base0);
            b.q[1] = *(const v4u*)(stg + sl * 32 + base0 + 16);
            // A rows all identical = sta chunk; same K split
            a.q[0] = *(const v4u*)(&sStaH[dBase + base0]);
            a.q[1] = *(const v4u*)(&sStaH[dBase + base0 + 16]);
            acc = __builtin_amdgcn_wmma_f32_16x16x32_f16(false, a.v, false, b.v,
                                                         (short)0, acc, false, false);
            __builtin_amdgcn_wave_barrier();
        }
        // D[m,n] identical over m (A rows equal); lanes 0..15 hold n=lane, m=0 in vgpr0
        if (lane < 16) sEu[sBase + lane] = acc[0];
    }
    __syncthreads();

    // ---- phase 2: p_eu / K0 = sum p*lp (|eu| small -> no max-shift needed) ----
    if (tid == 0) {
        float E = 0.f;
        for (int s = 0; s < N_TT; ++s) E += __expf(sEu[s]);
        float lE = __logf(E), invE = 1.0f / E, K0 = 0.f;
        for (int s = 0; s < N_TT; ++s) {
            float pz = __expf(sEu[s]) * invE;
            sPeu[s] = pz;
            K0 = fmaf(pz, sEu[s] - lE, K0);
        }
        sK0 = K0;
    }
    __syncthreads();

    // ---- phase 3: per-(c,p) column losses; ct in [-1.25,1.25], g in [-25,25]
    //      -> direct sum-exp is overflow-safe in f32, single fused pass ----
    const float K0 = sK0;
    const int tar = sTar;
    for (int j = tid; j < NCOL; j += 256) {
        const int p   = j & 7;
        const int cnc = sCnc[j];
        float se = 0.f, dot = 0.f;
        for (int s = 0; s < N_TT; ++s) {
            float cc = cg_cos(cnc, sPos[s * TPX + p]);
            float ct = (sCosSum[s] - sCosSp[s * TPX + p] + cc) * 0.125f;
            se += __expf(ct);
            dot = fmaf(sPeu[s], ct, dot);
        }
        out_dd[(size_t)t * NCOL + j] = K0 - dot + __logf(se);
        float se2 = 0.f, gt = 0.f;
        for (int k = 0; k < K_VOC; ++k) {
            const int s = N_TT + k;
            float cc = cg_cos(cnc, sPos[s * TPX + p]);
            float g  = (sCosSum[s] - sCosSp[s * TPX + p] + cc) * 2.5f;  // /TP * 20/TEMP*TEMP... = /8*20
            se2 += __expf(g);
            if (k == tar) gt = g;
        }
        out_ds[(size_t)t * NCOL + j] = __logf(se2) - gt;
    }
}

extern "C" void kernel_launch(void* const* d_in, const int* in_sizes, int n_in,
                              void* d_out, int out_size, void* d_ws, size_t ws_size,
                              hipStream_t stream) {
    (void)in_sizes; (void)n_in; (void)d_ws; (void)ws_size; (void)out_size;
    const int*   cur_tar = (const int*)d_in[0];
    const int*   cnc_loc = (const int*)d_in[1];
    const int*   sta_loc = (const int*)d_in[2];
    const int*   ttn_loc = (const int*)d_in[3];
    const int*   voc_loc = (const int*)d_in[4];
    const float* sta_emb = (const float*)d_in[5];
    const float* ttn_emb = (const float*)d_in[6];
    // d_in[7] (voc_emb) intentionally unused: eu_val[:, N_TTNBR:] is dead in the reference.
    float* out_dd = (float*)d_out;
    float* out_ds = out_dd + (size_t)T_CNT * NCOL;
    critigraph_kernel<<<dim3(T_CNT), dim3(256), 0, stream>>>(
        cur_tar, cnc_loc, sta_loc, ttn_loc, voc_loc, sta_emb, ttn_emb, out_dd, out_ds);
}
